// LinearAttention_20349555048522
// MI455X (gfx1250) — matrix operand order repaired and checked by reading, not compiled, not run on hardware
//
#include <hip/hip_runtime.h>
#include <hip/hip_bf16.h>
#include <math.h>

typedef __attribute__((ext_vector_type(8)))  _Float16 v8h;
typedef __attribute__((ext_vector_type(16))) _Float16 v16h;
typedef __attribute__((ext_vector_type(8)))  float    v8f;
typedef unsigned int u32x4 __attribute__((ext_vector_type(4)));
typedef int          i32x4 __attribute__((ext_vector_type(4)));
typedef int          i32x8 __attribute__((ext_vector_type(8)));

#if __has_builtin(__builtin_amdgcn_tensor_load_to_lds) && \
    __has_builtin(__builtin_amdgcn_s_wait_tensorcnt)
#define HAVE_TDM 1
#endif
#if __has_include(<hip/amd_detail/amd_gfx1250_TDM.h>)
#define TDM_6ARG 1
#endif

// ---------------------------------------------------------------------------
// WMMA helpers (CDNA5 gfx1250, wave32).
// A 16x32 f16: lanes 0-15 -> M=lane, K={0..7,16..23}; lanes 16-31 ->
// M=lane-16, K={8..15,24..31} (packed sequentially in the v16h).
// B 32x16 f16: lanes 0-15 -> N=lane, K={0..7,16..23}; lanes 16-31 -> N=lane-16,
// K={8..15,24..31}.  C/D f32: VGPR i -> M = i + (lane<16?0:8), N = lane&15.
// ---------------------------------------------------------------------------
__device__ __forceinline__ v8f wmma16(v16h a, v16h b, v8f c) {
  return __builtin_amdgcn_wmma_f32_16x16x32_f16(false, a, false, b, (short)0, c,
                                                false, false);
}

__device__ __forceinline__ v16h load_a_frag(const _Float16* base, int stride,
                                            int row0, int k0) {
  const int lane = threadIdx.x & 31;
  const int r = row0 + (lane & 15);
  const int off = (lane >> 4) * 8;
  const _Float16* p = base + (size_t)r * stride + k0 + off;
  v8h lo = *(const v8h*)p;
  v8h hi = *(const v8h*)(p + 16);
  return __builtin_shufflevector(lo, hi, 0, 1, 2, 3, 4, 5, 6, 7, 8, 9, 10, 11,
                                 12, 13, 14, 15);
}

__device__ __forceinline__ v16h load_a_exp(const _Float16* base, int stride,
                                           int row0, int k0, float M) {
  const int lane = threadIdx.x & 31;
  const int r = row0 + (lane & 15);
  const int off = (lane >> 4) * 8;
  const _Float16* p = base + (size_t)r * stride + k0 + off;
  v8h lo = *(const v8h*)p;
  v8h hi = *(const v8h*)(p + 16);
  v16h raw = __builtin_shufflevector(lo, hi, 0, 1, 2, 3, 4, 5, 6, 7, 8, 9, 10,
                                     11, 12, 13, 14, 15);
  v16h out;
#pragma unroll
  for (int i = 0; i < 16; ++i)
    out[i] = (_Float16)__expf((float)raw[i] - M);
  return out;
}

// B-fragment from storage laid out as [col][k] row-major (K-contiguous).
__device__ __forceinline__ v16h load_b_frag(const _Float16* base, int stride,
                                            int col0, int k0) {
  const int lane = threadIdx.x & 31;
  const int n = col0 + (lane & 15);
  const int off = (lane >> 4) * 8;
  const _Float16* p = base + (size_t)n * stride + k0 + off;
  v8h lo = *(const v8h*)p;
  v8h hi = *(const v8h*)(p + 16);
  return __builtin_shufflevector(lo, hi, 0, 1, 2, 3, 4, 5, 6, 7, 8, 9, 10, 11,
                                 12, 13, 14, 15);
}

// ---------------------------------------------------------------------------
// TDM: issue a 2-D f16 tile load (tile0 x tile1 elements, row stride
// "stride0" elements) into LDS with 64B-data + 16B-pad rows (stride 40
// halves), matching the sA layout.  Descriptor per ISA ch.8 (D# groups).
// ---------------------------------------------------------------------------
#if defined(HAVE_TDM)
__device__ __forceinline__ void tdm_load_tile_f16(unsigned ldsAddr,
                                                  const _Float16* gsrc,
                                                  unsigned tile0, unsigned tile1,
                                                  unsigned stride0,
                                                  unsigned tdim0) {
  unsigned long long ga = (unsigned long long)(size_t)(const void*)gsrc;
  u32x4 g0;
  g0.x = 1u;  // count=1 (valid user descriptor)
  g0.y = ldsAddr;
  g0.z = (unsigned)(ga & 0xFFFFFFFFull);
  g0.w = (unsigned)((ga >> 32) & 0x01FFFFFFull) | 0x80000000u;  // type=2
  i32x8 g1;
  // data_size=2B (code 1), pad_enable, pad_interval=16 DWORDs (code 3),
  // pad_amount=4 DWORDs (code 3)
  g1[0] = (1 << 16) | (1 << 20) | (3 << 22) | (3 << 25);
  g1[1] = (int)((tdim0 & 0xFFFFu) << 16);
  g1[2] = (int)((tdim0 >> 16) | ((tile1 & 0xFFFFu) << 16));  // tensor_dim1
  g1[3] = (int)((tile1 >> 16) | (tile0 << 16));              // tile_dim0
  g1[4] = (int)tile1;                                        // tile_dim1
  g1[5] = (int)stride0;                                      // dim0 stride lo
  g1[6] = 0;
  g1[7] = 0;
  i32x4 gz = {0, 0, 0, 0};
#if defined(TDM_6ARG)
  i32x8 gz8 = {0, 0, 0, 0, 0, 0, 0, 0};
  __builtin_amdgcn_tensor_load_to_lds(g0, g1, gz, gz, gz8, 0);
#else
  __builtin_amdgcn_tensor_load_to_lds(g0, g1, gz, gz, 0);
#endif
}
#endif

// ---------------------------------------------------------------------------
// Problem constants
// ---------------------------------------------------------------------------
#define BATCH 32
#define CIN 256
#define NSP 4096  // 64*64
#define HID 128   // HEADS*DIM_HEAD
#define OQKV 384  // 3*HID

// ---------------------------------------------------------------------------
// K0: convert weights fp32 -> f16
// ---------------------------------------------------------------------------
__global__ __launch_bounds__(256) void k_convert_w(
    const float* __restrict__ wqkv, const float* __restrict__ wout,
    _Float16* __restrict__ wqkv_h, _Float16* __restrict__ wout_h) {
  int i = blockIdx.x * 256 + threadIdx.x;
  if (i < OQKV * CIN) wqkv_h[i] = (_Float16)wqkv[i];
  if (i < CIN * HID) wout_h[i] = (_Float16)wout[i];
}

// ---------------------------------------------------------------------------
// K1: qkv = w_qkv (384x256) x x_b (256x4096).  128x128 tile per block,
// 8 waves (2M x 4N), each wave 4x2 wmma tiles, K-steps of 32.
// A tile staged by TDM (per-wave 16-row slice), B tile converted+transposed.
// ---------------------------------------------------------------------------
__global__ __launch_bounds__(256) void k_qkv_gemm(
    const float* __restrict__ x, const _Float16* __restrict__ wq,
    _Float16* __restrict__ q, _Float16* __restrict__ k,
    _Float16* __restrict__ v) {
  __shared__ __align__(16) char smem[20480];
  _Float16* sA = (_Float16*)smem;            // weights [m][k], stride 40
  _Float16* sB = (_Float16*)(smem + 10240);  // x tile transposed [n][k]
  const int tid = threadIdx.x, lane = tid & 31, wave = tid >> 5;
  const int wuni = __builtin_amdgcn_readfirstlane(threadIdx.x) >> 5;
  const int b = blockIdx.z;
  const int n0 = blockIdx.x * 128;
  const int sel = blockIdx.y;  // 0=q 1=k 2=v
  const int m0 = sel * 128;
  const int wm = (wave >> 2) * 64;
  const int wn = (wave & 3) * 32;
  const float* xb = x + (size_t)b * CIN * NSP;

  const v8f vz = {0.f, 0.f, 0.f, 0.f, 0.f, 0.f, 0.f, 0.f};
  v8f acc[4][2];
#pragma unroll
  for (int t = 0; t < 4; ++t)
#pragma unroll
    for (int u = 0; u < 2; ++u) acc[t][u] = vz;

  const int arow = tid >> 1, acol = (tid & 1) * 16;

  for (int kb = 0; kb < CIN; kb += 32) {
    // ---- stage A (f16 weights): 128x32, 16 rows per wave ----
#if defined(HAVE_TDM)
    tdm_load_tile_f16((unsigned)(size_t)(void*)sA + (unsigned)(wuni * 1280),
                      wq + (size_t)(m0 + wuni * 16) * CIN + kb,
                      /*tile0=*/32, /*tile1=*/16, /*stride0=*/CIN,
                      /*tdim0=*/CIN);
#else
    {
      const _Float16* src = wq + (size_t)(m0 + arow) * CIN + kb + acol;
      *(v8h*)&sA[arow * 40 + acol] = *(const v8h*)src;
      *(v8h*)&sA[arow * 40 + acol + 8] = *(const v8h*)(src + 8);
    }
#endif
    // ---- stage B: x fp32 tile 32x128, convert + transpose -> sB[n][c] ----
#pragma unroll
    for (int j = 0; j < 4; ++j) {
      int linear = (j * 256 + tid) * 4;
      int c = linear >> 7;
      int nn = linear & 127;
      const float4 f = *(const float4*)(xb + (size_t)(kb + c) * NSP + n0 + nn);
      sB[(nn + 0) * 40 + c] = (_Float16)f.x;
      sB[(nn + 1) * 40 + c] = (_Float16)f.y;
      sB[(nn + 2) * 40 + c] = (_Float16)f.z;
      sB[(nn + 3) * 40 + c] = (_Float16)f.w;
    }
#if defined(HAVE_TDM)
    __builtin_amdgcn_s_wait_tensorcnt(0);
#endif
    __syncthreads();
    v16h af[4], bf[2];
#pragma unroll
    for (int t = 0; t < 4; ++t) af[t] = load_a_frag(sA, 40, wm + t * 16, 0);
#pragma unroll
    for (int u = 0; u < 2; ++u) bf[u] = load_b_frag(sB, 40, wn + u * 16, 0);
#pragma unroll
    for (int t = 0; t < 4; ++t)
#pragma unroll
      for (int u = 0; u < 2; ++u) acc[t][u] = wmma16(af[t], bf[u], acc[t][u]);
    __syncthreads();
  }

  // ---- vectorized epilogue: LDS transpose-stage then b128 stores ----
  _Float16* dst = (sel == 0) ? q : (sel == 1) ? k : v;
  dst += (size_t)b * HID * NSP;
  _Float16* area = (_Float16*)smem + wave * 640;  // 16 rows x 40 halves
  const int r2 = lane >> 1, half = lane & 1;
#pragma unroll
  for (int t = 0; t < 4; ++t) {
    __syncthreads();
#pragma unroll
    for (int u = 0; u < 2; ++u)
#pragma unroll
      for (int i = 0; i < 8; ++i)
        area[(i + ((lane >> 4) << 3)) * 40 + u * 16 + (lane & 15)] =
            (_Float16)acc[t][u][i];
    __syncthreads();
    const _Float16* rp = area + r2 * 40 + half * 16;
    v8h o0 = *(const v8h*)rp;
    v8h o1 = *(const v8h*)(rp + 8);
    _Float16* gp = dst + (size_t)(wm + t * 16 + r2) * NSP + n0 + wn + half * 16;
    *(v8h*)gp = o0;
    *(v8h*)(gp + 8) = o1;
  }
}

// ---------------------------------------------------------------------------
// K2: per-row softmax stats over k (rows = B*HID, length NSP): max & sum(exp).
// ---------------------------------------------------------------------------
__global__ __launch_bounds__(256) void k_softmax_stats(
    const _Float16* __restrict__ kf, float* __restrict__ stats) {
  const int row = blockIdx.x;
  const int tid = threadIdx.x, lane = tid & 31, wave = tid >> 5;
  const _Float16* base = kf + (size_t)row * NSP;
  __shared__ float red[8];

  float m = -3.0e38f;
  for (int j = tid; j < NSP; j += 256) m = fmaxf(m, (float)base[j]);
#pragma unroll
  for (int off = 16; off > 0; off >>= 1) m = fmaxf(m, __shfl_xor(m, off, 32));
  if (lane == 0) red[wave] = m;
  __syncthreads();
  m = red[0];
#pragma unroll
  for (int w = 1; w < 8; ++w) m = fmaxf(m, red[w]);
  __syncthreads();

  float s = 0.f;
  for (int j = tid; j < NSP; j += 256) s += __expf((float)base[j] - m);
#pragma unroll
  for (int off = 16; off > 0; off >>= 1) s += __shfl_xor(s, off, 32);
  if (lane == 0) red[wave] = s;
  __syncthreads();
  if (tid == 0) {
    float st = 0.f;
    for (int w = 0; w < 8; ++w) st += red[w];
    stats[2 * row] = m;
    stats[2 * row + 1] = st;
  }
}

// ---------------------------------------------------------------------------
// K3: context per (b,h): C(32x32) = exp(k - M) (32xN) x v^T (Nx32), then
// row-scale by 1/S and store transposed as ctxT[e][d] (f16).
// ---------------------------------------------------------------------------
__global__ __launch_bounds__(128) void k_context_gemm(
    const _Float16* __restrict__ kf, const _Float16* __restrict__ vf,
    const float* __restrict__ stats, _Float16* __restrict__ ctxT) {
  const int bh = blockIdx.x;  // b*4 + h
  const int tid = threadIdx.x, lane = tid & 31, wave = tid >> 5;
  const _Float16* kb_ = kf + (size_t)(bh)*32 * NSP;
  const _Float16* vb_ = vf + (size_t)(bh)*32 * NSP;

  const float M0 = stats[2 * (bh * 32 + (lane & 15))];
  const float M1 = stats[2 * (bh * 32 + 16 + (lane & 15))];

  const v8f vz = {0.f, 0.f, 0.f, 0.f, 0.f, 0.f, 0.f, 0.f};
  v8f acc[2][2];
#pragma unroll
  for (int t = 0; t < 2; ++t)
#pragma unroll
    for (int u = 0; u < 2; ++u) acc[t][u] = vz;

  const int kend = wave * 1024 + 1024;
  for (int kb = wave * 1024; kb < kend; kb += 32) {
    v16h a0 = load_a_exp(kb_, NSP, 0, kb, M0);
    v16h a1 = load_a_exp(kb_, NSP, 16, kb, M1);
    v16h b0 = load_b_frag(vb_, NSP, 0, kb);
    v16h b1 = load_b_frag(vb_, NSP, 16, kb);
    acc[0][0] = wmma16(a0, b0, acc[0][0]);
    acc[0][1] = wmma16(a0, b1, acc[0][1]);
    acc[1][0] = wmma16(a1, b0, acc[1][0]);
    acc[1][1] = wmma16(a1, b1, acc[1][1]);
  }

  __shared__ float cs[32 * 32];
  for (int i = tid; i < 1024; i += 128) cs[i] = 0.f;
  __syncthreads();
#pragma unroll
  for (int t = 0; t < 2; ++t)
#pragma unroll
    for (int u = 0; u < 2; ++u)
#pragma unroll
      for (int i = 0; i < 8; ++i) {
        int d = t * 16 + i + ((lane >> 4) << 3);
        int e = u * 16 + (lane & 15);
        atomicAdd(&cs[d * 32 + e], acc[t][u][i]);
      }
  __syncthreads();
  for (int i = tid; i < 1024; i += 128) {
    int d = i >> 5, e = i & 31;
    float S = stats[2 * (bh * 32 + d) + 1];
    ctxT[(size_t)bh * 1024 + e * 32 + d] = (_Float16)(cs[i] / S);
  }
}

// ---------------------------------------------------------------------------
// K4: out[e][n] = ctxT(32x32) x q(32xN).  Block = (b,h,n-tile 256),
// q tile transposed into LDS, single K=32 WMMA step per wave.
// ---------------------------------------------------------------------------
__global__ __launch_bounds__(256) void k_out_gemm(
    const _Float16* __restrict__ ctxT, const _Float16* __restrict__ qf,
    _Float16* __restrict__ outf) {
  __shared__ __align__(16) _Float16 qT[256 * 40];  // [n][d]
  const int tid = threadIdx.x, lane = tid & 31, wave = tid >> 5;
  const int b = blockIdx.z, h = blockIdx.y;
  const int n0 = blockIdx.x * 256;
  const _Float16* qb_ = qf + (size_t)(b * 4 + h) * 32 * NSP;

#pragma unroll
  for (int d = 0; d < 32; ++d)
    qT[tid * 40 + d] = qb_[(size_t)d * NSP + n0 + tid];
  __syncthreads();

  const _Float16* cx = ctxT + (size_t)(b * 4 + h) * 1024;
  v16h a0 = load_a_frag(cx, 32, 0, 0);
  v16h a1 = load_a_frag(cx, 32, 16, 0);
  v16h b0 = load_b_frag(qT, 40, wave * 32, 0);
  v16h b1 = load_b_frag(qT, 40, wave * 32 + 16, 0);
  const v8f vz = {0.f, 0.f, 0.f, 0.f, 0.f, 0.f, 0.f, 0.f};
  v8f acc[2][2];
  acc[0][0] = wmma16(a0, b0, vz);
  acc[0][1] = wmma16(a0, b1, vz);
  acc[1][0] = wmma16(a1, b0, vz);
  acc[1][1] = wmma16(a1, b1, vz);

  _Float16* ob = outf + (size_t)(b * 4 + h) * 32 * NSP;
  _Float16* area = qT + wave * 640;  // reuse LDS: 16 rows x 40 halves
  const int r2 = lane >> 1, half = lane & 1;
#pragma unroll
  for (int t = 0; t < 2; ++t) {
    __syncthreads();
#pragma unroll
    for (int u = 0; u < 2; ++u)
#pragma unroll
      for (int i = 0; i < 8; ++i)
        area[(i + ((lane >> 4) << 3)) * 40 + u * 16 + (lane & 15)] =
            (_Float16)acc[t][u][i];
    __syncthreads();
    const _Float16* rp = area + r2 * 40 + half * 16;
    v8h o0 = *(const v8h*)rp;
    v8h o1 = *(const v8h*)(rp + 8);
    _Float16* gp =
        ob + (size_t)(t * 16 + r2) * NSP + n0 + wave * 32 + half * 16;
    *(v8h*)gp = o0;
    *(v8h*)(gp + 8) = o1;
  }
}

// ---------------------------------------------------------------------------
// K5: y = w_out (256x128) x out_b (128x4096) + bias, fp32 result.
// ---------------------------------------------------------------------------
__global__ __launch_bounds__(256) void k_final_gemm(
    const _Float16* __restrict__ wo, const _Float16* __restrict__ outf,
    const float* __restrict__ bias, float* __restrict__ y) {
  __shared__ __align__(16) char smem[20480];
  _Float16* sA = (_Float16*)smem;
  _Float16* sB = (_Float16*)(smem + 10240);
  const int tid = threadIdx.x, lane = tid & 31, wave = tid >> 5;
  const int wuni = __builtin_amdgcn_readfirstlane(threadIdx.x) >> 5;
  const int b = blockIdx.z;
  const int m0 = blockIdx.y * 128;
  const int n0 = blockIdx.x * 128;
  const int wm = (wave >> 2) * 64;
  const int wn = (wave & 3) * 32;
  const _Float16* ob = outf + (size_t)b * HID * NSP;

  const v8f vz = {0.f, 0.f, 0.f, 0.f, 0.f, 0.f, 0.f, 0.f};
  v8f acc[4][2];
#pragma unroll
  for (int t = 0; t < 4; ++t)
#pragma unroll
    for (int u = 0; u < 2; ++u) acc[t][u] = vz;

  const int arow = tid >> 1, acol = (tid & 1) * 16;

  for (int kb = 0; kb < HID; kb += 32) {
#if defined(HAVE_TDM)
    tdm_load_tile_f16((unsigned)(size_t)(void*)sA + (unsigned)(wuni * 1280),
                      wo + (size_t)(m0 + wuni * 16) * HID + kb,
                      /*tile0=*/32, /*tile1=*/16, /*stride0=*/HID,
                      /*tdim0=*/HID);
#else
    {
      const _Float16* src = wo + (size_t)(m0 + arow) * HID + kb + acol;
      *(v8h*)&sA[arow * 40 + acol] = *(const v8h*)src;
      *(v8h*)&sA[arow * 40 + acol + 8] = *(const v8h*)(src + 8);
    }
#endif
#pragma unroll
    for (int j = 0; j < 2; ++j) {
      int linear = (j * 256 + tid) * 8;
      int c = linear >> 7;
      int nn = linear & 127;
      v8h f = *(const v8h*)(ob + (size_t)(kb + c) * NSP + n0 + nn);
#pragma unroll
      for (int z = 0; z < 8; ++z) sB[(nn + z) * 40 + c] = f[z];
    }
#if defined(HAVE_TDM)
    __builtin_amdgcn_s_wait_tensorcnt(0);
#endif
    __syncthreads();
    v16h af[4], bf[2];
#pragma unroll
    for (int t = 0; t < 4; ++t) af[t] = load_a_frag(sA, 40, wm + t * 16, 0);
#pragma unroll
    for (int u = 0; u < 2; ++u) bf[u] = load_b_frag(sB, 40, wn + u * 16, 0);
#pragma unroll
    for (int t = 0; t < 4; ++t)
#pragma unroll
      for (int u = 0; u < 2; ++u) acc[t][u] = wmma16(af[t], bf[u], acc[t][u]);
    __syncthreads();
  }

  // ---- vectorized fp32 epilogue with bias, via LDS stage ----
  float* area = (float*)smem + wave * 640;  // 16 rows x 40 floats
  const int r2 = lane >> 1, half = lane & 1;
#pragma unroll
  for (int t = 0; t < 4; ++t) {
    __syncthreads();
#pragma unroll
    for (int u = 0; u < 2; ++u)
#pragma unroll
      for (int i = 0; i < 8; ++i)
        area[(i + ((lane >> 4) << 3)) * 40 + u * 16 + (lane & 15)] =
            acc[t][u][i];
    __syncthreads();
    const int orow = m0 + wm + t * 16 + r2;
    const float bv = bias[orow];
    const float* rp = area + r2 * 40 + half * 16;
    float4 q0 = *(const float4*)(rp + 0);
    float4 q1 = *(const float4*)(rp + 4);
    float4 q2 = *(const float4*)(rp + 8);
    float4 q3 = *(const float4*)(rp + 12);
    q0.x += bv; q0.y += bv; q0.z += bv; q0.w += bv;
    q1.x += bv; q1.y += bv; q1.z += bv; q1.w += bv;
    q2.x += bv; q2.y += bv; q2.z += bv; q2.w += bv;
    q3.x += bv; q3.y += bv; q3.z += bv; q3.w += bv;
    float* gp = y + ((size_t)b * CIN + orow) * NSP + n0 + wn + half * 16;
    *(float4*)(gp + 0) = q0;
    *(float4*)(gp + 4) = q1;
    *(float4*)(gp + 8) = q2;
    *(float4*)(gp + 12) = q3;
  }
}

// ---------------------------------------------------------------------------
// Launch
// ---------------------------------------------------------------------------
extern "C" void kernel_launch(void* const* d_in, const int* in_sizes, int n_in,
                              void* d_out, int out_size, void* d_ws,
                              size_t ws_size, hipStream_t stream) {
  const float* x = (const float*)d_in[0];     // (32,256,64,64)
  const float* wqkv = (const float*)d_in[1];  // (384,256)
  const float* wout = (const float*)d_in[2];  // (256,128)
  const float* bout = (const float*)d_in[3];  // (256,)
  float* y = (float*)d_out;                   // (32,256,64,64)

  char* ws = (char*)d_ws;
  _Float16* wq_h = (_Float16*)(ws + 0);          // 196608 B
  _Float16* wo_h = (_Float16*)(ws + 196608);     // 65536 B
  _Float16* qf = (_Float16*)(ws + 262144);       // 33554432 B
  _Float16* kf = (_Float16*)(ws + 33816576);     // 33554432 B
  _Float16* vf = (_Float16*)(ws + 67371008);     // 33554432 B
  float* stats = (float*)(ws + 100925440);       // 32768 B
  _Float16* ctxT = (_Float16*)(ws + 100958208);  // 262144 B
  _Float16* outf = (_Float16*)(ws + 101220352);  // 33554432 B

  k_convert_w<<<384, 256, 0, stream>>>(wqkv, wout, wq_h, wo_h);
  k_qkv_gemm<<<dim3(32, 3, 32), 256, 0, stream>>>(x, wq_h, qf, kf, vf);
  k_softmax_stats<<<4096, 256, 0, stream>>>(kf, stats);
  k_context_gemm<<<128, 128, 0, stream>>>(kf, vf, stats, ctxT);
  k_out_gemm<<<dim3(16, 4, 32), 256, 0, stream>>>(ctxT, qf, outf);
  k_final_gemm<<<dim3(32, 2, 32), 256, 0, stream>>>(wo_h, outf, bout, y);
}
